// RSVDrugScreeningModel_42958262895196
// MI455X (gfx1250) — matrix operand order repaired and checked by reading, not compiled, not run on hardware
//
#include <hip/hip_runtime.h>
#include <hip/hip_bf16.h>
#include <math.h>

#define HID     128
#define NUM_RBF 64
#define NNODES  (128 * 80)
#define PI_F    3.14159265358979f

typedef __attribute__((ext_vector_type(16))) _Float16 v16h;
typedef __attribute__((ext_vector_type(8)))  float    v8f;

__device__ __forceinline__ v8f wmma16(v16h a, v16h b, v8f c) {
  return __builtin_amdgcn_wmma_f32_16x16x32_f16(false, a, false, b, (short)0, c, false, false);
}

// ---------------------------------------------------------------------------
// GEMM helper: [16 x (KS*32)] f16 tile in LDS  x  [K x 128] fragment-ordered
// f16 weights in global (L2-resident). Each wave computes two 16x16 output
// column tiles (ct = 2w, 2w+1) so two independent WMMA chains interleave.
// A-fragment lane layout per ISA 7.12.2 (16-bit A, 16x32):
//   lane l (row M = l&15) holds K = ks*32 + (h/8)*16 + 8*(l>=16) + (h&7)
// B fragments pre-swizzled so each lane loads 16 contiguous halves.
// ---------------------------------------------------------------------------
template<int KS>
__device__ __forceinline__ void gemm_2ct(const _Float16* As, int astride,
                                         const _Float16* __restrict__ bf,
                                         const float* __restrict__ bias,
                                         int lane, int w, v8f& acc0, v8f& acc1) {
  const int nlo = lane & 15;
  const int hiK = (lane >= 16) ? 8 : 0;
  const int ct0 = 2 * w, ct1 = 2 * w + 1;
  const float bv0 = bias[ct0 * 16 + nlo];
  const float bv1 = bias[ct1 * 16 + nlo];
#pragma unroll
  for (int r = 0; r < 8; ++r) { acc0[r] = bv0; acc1[r] = bv1; }
  for (int ks = 0; ks < KS; ++ks) {
    v16h a;
    const _Float16* ap = As + nlo * astride + ks * 32 + hiK;
#pragma unroll
    for (int h = 0; h < 16; ++h) a[h] = ap[((h >> 3) << 4) + (h & 7)];
    v16h b0 = *(const v16h*)(bf + (size_t)((ct0 * KS + ks) * 32 + lane) * 16);
    v16h b1 = *(const v16h*)(bf + (size_t)((ct1 * KS + ks) * 32 + lane) * 16);
    acc0 = wmma16(a, b0, acc0);
    acc1 = wmma16(a, b1, acc1);
  }
}

// SiLU on the two accumulator tiles, store f16 into LDS [16 x HID] (stride hs).
// C/D layout: element r of lane l is (M = r + 8*(l>=16), N = ct*16 + (l&15)).
__device__ __forceinline__ void silu_store(v8f acc0, v8f acc1, _Float16* H,
                                           int hs, int lane, int w) {
  const int nlo = lane & 15;
  const int moff = (lane >= 16) ? 8 : 0;
#pragma unroll
  for (int r = 0; r < 8; ++r) {
    float v0 = acc0[r]; v0 = v0 / (1.0f + __expf(-v0));
    float v1 = acc1[r]; v1 = v1 / (1.0f + __expf(-v1));
    H[(r + moff) * hs + (2 * w)     * 16 + nlo] = (_Float16)v0;
    H[(r + moff) * hs + (2 * w + 1) * 16 + nlo] = (_Float16)v1;
  }
}

// ---------------------------------------------------------------------------
// Small utility kernels
// ---------------------------------------------------------------------------
__global__ void k_zero(float* p, int n) {
  for (int i = blockIdx.x * blockDim.x + threadIdx.x; i < n; i += gridDim.x * blockDim.x)
    p[i] = 0.0f;
}

__global__ void k_xh(const float* __restrict__ x, _Float16* __restrict__ xh, int n) {
  for (int i = blockIdx.x * blockDim.x + threadIdx.x; i < n; i += gridDim.x * blockDim.x)
    xh[i] = (_Float16)x[i];
}

// Swizzle f32 [Ksrc x 128] row-major weights into fragment-ordered f16
// (zero-padded to Kpad rows). dst index = ((ct*KS + ks)*32 + lane)*16 + h
// with K = ks*32 + (h/8)*16 + 8*(lane>=16) + (h&7), N = ct*16 + (lane&15).
__global__ void k_wfrag(const float* __restrict__ src, _Float16* __restrict__ dst,
                        int Ksrc, int Kpad) {
  const int KS = Kpad >> 5;
  const int total = Kpad * HID;
  for (int idx = blockIdx.x * blockDim.x + threadIdx.x; idx < total;
       idx += gridDim.x * blockDim.x) {
    int h    = idx & 15;
    int lane = (idx >> 4) & 31;
    int ks   = (idx >> 9) % KS;
    int ct   = (idx >> 9) / KS;
    int k = ks * 32 + ((h >> 3) << 4) + ((lane >= 16) ? 8 : 0) + (h & 7);
    int n = ct * 16 + (lane & 15);
    float v = (k < Ksrc) ? src[k * HID + n] : 0.0f;
    dst[idx] = (_Float16)v;
  }
}

__global__ void k_final(float* __restrict__ acc, const float* __restrict__ cnt, int total) {
  for (int i = blockIdx.x * blockDim.x + threadIdx.x; i < total; i += gridDim.x * blockDim.x) {
    float c = cnt[i >> 7];
    acc[i] = acc[i] / fmaxf(c, 1.0f);
  }
}

// ---------------------------------------------------------------------------
// Per-scale message kernel: one block = one tile of 16 edges, 4 waves.
// GEMM1: [16x320] @ [320x128]  (10 k-steps)   GEMM2: [16x128] @ [128x128]
// ---------------------------------------------------------------------------
__global__ __launch_bounds__(128) void k_msg(
    const _Float16* __restrict__ xh, const float* __restrict__ pos,
    const int* __restrict__ row, const int* __restrict__ col, int E,
    const _Float16* __restrict__ w1f, const float* __restrict__ b1,
    const _Float16* __restrict__ w2f, const float* __restrict__ b2,
    float* __restrict__ acc, float* __restrict__ cnt, float cut) {
  __shared__ __align__(16) _Float16 Asm[16][328];   // 320 + pad
  __shared__ __align__(16) _Float16 Hsm[16][136];
  __shared__ int   s_r[16];
  __shared__ int   s_rc[32];
  __shared__ float s_cw[16];
  __shared__ float s_d[16];
  const int tid = threadIdx.x;
  const int e0 = blockIdx.x * 16;

  if (tid < 16) {
    int e = e0 + tid;
    int r = 0, c = 0; float cw = 0.f, d = 0.f;
    if (e < E) {
      r = row[e]; c = col[e];
      float dx = pos[c * 3 + 0] - pos[r * 3 + 0];
      float dy = pos[c * 3 + 1] - pos[r * 3 + 1];
      float dz = pos[c * 3 + 2] - pos[r * 3 + 2];
      d = sqrtf(dx * dx + dy * dy + dz * dz);
      cw = (d < cut) ? 0.5f * (__cosf(d * PI_F / cut) + 1.0f) : 0.0f;
      atomicAdd(&cnt[r], 1.0f);
      s_r[tid] = r;
    } else {
      s_r[tid] = -1;
    }
    s_cw[tid] = cw; s_d[tid] = d;
    s_rc[tid] = r;  s_rc[16 + tid] = c;
  }
  __syncthreads();

  // gather x[row] | x[col] : 16 edges x 32 chunks of 8 halves (uint4)
  for (int i = tid; i < 512; i += 128) {
    int m = i >> 5, ch = i & 31;
    int node = (ch < 16) ? s_rc[m] : s_rc[16 + m];
    int off = (ch & 15) * 8;
    uint4 v = *(const uint4*)(xh + (size_t)node * HID + off);
    int dsto = (ch < 16) ? off : (HID + off);
    *(uint4*)(&Asm[m][dsto]) = v;
  }
  // RBF features
  const float wg = cut * (0.5f / (float)NUM_RBF);
  const float inv2w2 = 1.0f / (2.0f * wg * wg);
  for (int i = tid; i < 16 * NUM_RBF; i += 128) {
    int m = i >> 6, j = i & 63;
    float center = cut * (float)j * (1.0f / 63.0f);
    float dd = s_d[m] - center;
    Asm[m][256 + j] = (_Float16)__expf(-dd * dd * inv2w2);
  }
  __syncthreads();

  const int lane = tid & 31, w = tid >> 5;
  v8f acc0, acc1;
  gemm_2ct<10>(&Asm[0][0], 328, w1f, b1, lane, w, acc0, acc1);
  silu_store(acc0, acc1, &Hsm[0][0], 136, lane, w);
  __syncthreads();
  gemm_2ct<4>(&Hsm[0][0], 136, w2f, b2, lane, w, acc0, acc1);

  const int nlo = lane & 15, moff = (lane >= 16) ? 8 : 0;
#pragma unroll
  for (int r = 0; r < 8; ++r) {
    int m = r + moff;
    int rn = s_r[m];
    if (rn >= 0) {
      float cw = s_cw[m];
      atomicAdd(&acc[(size_t)rn * HID + (2 * w)     * 16 + nlo], acc0[r] * cw);
      atomicAdd(&acc[(size_t)rn * HID + (2 * w + 1) * 16 + nlo], acc1[r] * cw);
    }
  }
}

// ---------------------------------------------------------------------------
// Bond kernel: [16x288(=262 padded)] @ [288x128] then [16x128] @ [128x128]
// ---------------------------------------------------------------------------
__global__ __launch_bounds__(128) void k_bond(
    const _Float16* __restrict__ xh, const float* __restrict__ attr,
    const int* __restrict__ row, const int* __restrict__ col, int E,
    const _Float16* __restrict__ w1f, const float* __restrict__ b1,
    const _Float16* __restrict__ w2f, const float* __restrict__ b2,
    float* __restrict__ acc, float* __restrict__ cnt) {
  __shared__ __align__(16) _Float16 Asm[16][296];   // 288 + pad
  __shared__ __align__(16) _Float16 Hsm[16][136];
  __shared__ int s_r[16];
  __shared__ int s_rc[32];
  const int tid = threadIdx.x;
  const int e0 = blockIdx.x * 16;

  if (tid < 16) {
    int e = e0 + tid;
    int r = 0, c = 0;
    if (e < E) {
      r = row[e]; c = col[e];
      atomicAdd(&cnt[r], 1.0f);
      s_r[tid] = r;
    } else {
      s_r[tid] = -1;
    }
    s_rc[tid] = r; s_rc[16 + tid] = c;
  }
  __syncthreads();
  for (int i = tid; i < 512; i += 128) {
    int m = i >> 5, ch = i & 31;
    int node = (ch < 16) ? s_rc[m] : s_rc[16 + m];
    int off = (ch & 15) * 8;
    uint4 v = *(const uint4*)(xh + (size_t)node * HID + off);
    int dsto = (ch < 16) ? off : (HID + off);
    *(uint4*)(&Asm[m][dsto]) = v;
  }
  for (int i = tid; i < 16 * 32; i += 128) {      // cols 256..287: attr + zero pad
    int m = i >> 5, j = i & 31;
    int e = e0 + m;
    float v = (e < E && j < 6) ? attr[(size_t)e * 6 + j] : 0.0f;
    Asm[m][256 + j] = (_Float16)v;
  }
  __syncthreads();

  const int lane = tid & 31, w = tid >> 5;
  v8f acc0, acc1;
  gemm_2ct<9>(&Asm[0][0], 296, w1f, b1, lane, w, acc0, acc1);
  silu_store(acc0, acc1, &Hsm[0][0], 136, lane, w);
  __syncthreads();
  gemm_2ct<4>(&Hsm[0][0], 136, w2f, b2, lane, w, acc0, acc1);

  const int nlo = lane & 15, moff = (lane >= 16) ? 8 : 0;
#pragma unroll
  for (int r = 0; r < 8; ++r) {
    int m = r + moff;
    int rn = s_r[m];
    if (rn >= 0) {
      atomicAdd(&acc[(size_t)rn * HID + (2 * w)     * 16 + nlo], acc0[r]);
      atomicAdd(&acc[(size_t)rn * HID + (2 * w + 1) * 16 + nlo], acc1[r]);
    }
  }
}

// ---------------------------------------------------------------------------
// Fused attention + update: one block = 16 nodes.
// attn: [16x768]@[768x128] SiLU, 128x6 logits on VALU, softmax,
// weighted sum, upd: [16x256]@[256x128] SiLU @[128x128], residual out.
// ---------------------------------------------------------------------------
__global__ __launch_bounds__(128) void k_attn_upd(
    const float* __restrict__ x, const _Float16* __restrict__ xh,
    const float* __restrict__ accall,
    const _Float16* __restrict__ aw1f, const float* __restrict__ ab1,
    const float* __restrict__ aw2, const float* __restrict__ ab2,
    const _Float16* __restrict__ uw1f, const float* __restrict__ ub1,
    const _Float16* __restrict__ uw2f, const float* __restrict__ ub2,
    float* __restrict__ out) {
  __shared__ __align__(16) _Float16 Aat[16][776];   // 768 + pad
  __shared__ __align__(16) _Float16 Hsm[16][136];
  __shared__ __align__(16) _Float16 Aup[16][264];   // 256 + pad
  __shared__ float Lg[16][6];
  __shared__ float At[16][6];
  const int tid = threadIdx.x;
  const int lane = tid & 31, w = tid >> 5;
  const int n0 = blockIdx.x * 16;
  const size_t NH = (size_t)NNODES * HID;

  for (int i = tid; i < 16 * 768; i += 128) {
    int m = i / 768, f = i % 768;
    int s = f >> 7, n = f & 127;
    Aat[m][f] = (_Float16)accall[(size_t)s * NH + (size_t)(n0 + m) * HID + n];
  }
  __syncthreads();

  v8f acc0, acc1;
  gemm_2ct<24>(&Aat[0][0], 776, aw1f, ab1, lane, w, acc0, acc1);
  silu_store(acc0, acc1, &Hsm[0][0], 136, lane, w);
  __syncthreads();

  if (tid < 96) {                                   // 16 nodes x 6 logits
    int m = tid / 6, s = tid % 6;
    float sum = ab2[s];
    for (int k = 0; k < HID; ++k) sum += (float)Hsm[m][k] * aw2[k * 6 + s];
    Lg[m][s] = sum;
  }
  __syncthreads();
  if (tid < 16) {
    float mx = Lg[tid][0];
    for (int s = 1; s < 6; ++s) mx = fmaxf(mx, Lg[tid][s]);
    float e[6], ssum = 0.f;
    for (int s = 0; s < 6; ++s) { e[s] = __expf(Lg[tid][s] - mx); ssum += e[s]; }
    for (int s = 0; s < 6; ++s) At[tid][s] = e[s] / ssum;
  }
  __syncthreads();

  for (int i = tid; i < 16 * HID; i += 128) {       // weighted sum + build upd input
    int m = i >> 7, n = i & 127;
    float sum = 0.f;
    for (int s = 0; s < 6; ++s)
      sum += accall[(size_t)s * NH + (size_t)(n0 + m) * HID + n] * At[m][s];
    Aup[m][HID + n] = (_Float16)sum;
    Aup[m][n] = xh[(size_t)(n0 + m) * HID + n];
  }
  __syncthreads();

  gemm_2ct<8>(&Aup[0][0], 264, uw1f, ub1, lane, w, acc0, acc1);
  silu_store(acc0, acc1, &Hsm[0][0], 136, lane, w);
  __syncthreads();
  gemm_2ct<4>(&Hsm[0][0], 136, uw2f, ub2, lane, w, acc0, acc1);

  const int nlo = lane & 15, moff = (lane >= 16) ? 8 : 0;
#pragma unroll
  for (int r = 0; r < 8; ++r) {
    int m = r + moff;
    size_t o0 = (size_t)(n0 + m) * HID + (2 * w)     * 16 + nlo;
    size_t o1 = (size_t)(n0 + m) * HID + (2 * w + 1) * 16 + nlo;
    out[o0] = x[o0] + acc0[r];
    out[o1] = x[o1] + acc1[r];
  }
}

// ---------------------------------------------------------------------------
extern "C" void kernel_launch(void* const* d_in, const int* in_sizes, int n_in,
                              void* d_out, int out_size, void* d_ws, size_t ws_size,
                              hipStream_t stream) {
  (void)n_in; (void)out_size; (void)ws_size;
  const float* x        = (const float*)d_in[0];
  const float* pos      = (const float*)d_in[1];
  const float* battr    = (const float*)d_in[2];
  const float* msg_w1   = (const float*)d_in[3];
  const float* msg_b1   = (const float*)d_in[4];
  const float* msg_w2   = (const float*)d_in[5];
  const float* msg_b2   = (const float*)d_in[6];
  const float* bond_w1  = (const float*)d_in[7];
  const float* bond_b1  = (const float*)d_in[8];
  const float* bond_w2  = (const float*)d_in[9];
  const float* bond_b2  = (const float*)d_in[10];
  const float* attn_w1  = (const float*)d_in[11];
  const float* attn_b1  = (const float*)d_in[12];
  const float* attn_w2  = (const float*)d_in[13];
  const float* attn_b2  = (const float*)d_in[14];
  const float* upd_w1   = (const float*)d_in[15];
  const float* upd_b1   = (const float*)d_in[16];
  const float* upd_w2   = (const float*)d_in[17];
  const float* upd_b2   = (const float*)d_in[18];

  // workspace carve-up (256B aligned)
  char* ws = (char*)d_ws;
  size_t off = 0;
  auto carve = [&](size_t bytes) {
    off = (off + 255) & ~(size_t)255;
    size_t o = off; off += bytes; return o;
  };
  _Float16* xh       = (_Float16*)(ws + carve((size_t)NNODES * HID * 2));
  _Float16* w1f_msg  = (_Float16*)(ws + carve((size_t)5 * 320 * HID * 2));
  _Float16* w2f_msg  = (_Float16*)(ws + carve((size_t)5 * 128 * HID * 2));
  _Float16* w1f_bond = (_Float16*)(ws + carve((size_t)288 * HID * 2));
  _Float16* w2f_bond = (_Float16*)(ws + carve((size_t)128 * HID * 2));
  _Float16* w1f_attn = (_Float16*)(ws + carve((size_t)768 * HID * 2));
  _Float16* w1f_upd  = (_Float16*)(ws + carve((size_t)256 * HID * 2));
  _Float16* w2f_upd  = (_Float16*)(ws + carve((size_t)128 * HID * 2));
  float*    accs     = (float*)(ws + carve((size_t)6 * NNODES * HID * 4));
  float*    cnts     = (float*)(ws + carve((size_t)6 * NNODES * 4));

  const int TPB = 256;
  // 1. zero accumulators + counts
  {
    int nz = 6 * NNODES * HID;
    k_zero<<<(nz + TPB - 1) / TPB, TPB, 0, stream>>>(accs, nz);
    int nc = 6 * NNODES;
    k_zero<<<(nc + TPB - 1) / TPB, TPB, 0, stream>>>(cnts, nc);
  }
  // 2. x -> f16
  {
    int n = NNODES * HID;
    k_xh<<<(n + TPB - 1) / TPB, TPB, 0, stream>>>(x, xh, n);
  }
  // 3. weight swizzles
  for (int j = 0; j < 5; ++j) {
    k_wfrag<<<(320 * HID + TPB - 1) / TPB, TPB, 0, stream>>>(
        msg_w1 + (size_t)j * 320 * HID, w1f_msg + (size_t)j * 320 * HID, 320, 320);
    k_wfrag<<<(128 * HID + TPB - 1) / TPB, TPB, 0, stream>>>(
        msg_w2 + (size_t)j * 128 * HID, w2f_msg + (size_t)j * 128 * HID, 128, 128);
  }
  k_wfrag<<<(288 * HID + TPB - 1) / TPB, TPB, 0, stream>>>(bond_w1, w1f_bond, 262, 288);
  k_wfrag<<<(128 * HID + TPB - 1) / TPB, TPB, 0, stream>>>(bond_w2, w2f_bond, 128, 128);
  k_wfrag<<<(768 * HID + TPB - 1) / TPB, TPB, 0, stream>>>(attn_w1, w1f_attn, 768, 768);
  k_wfrag<<<(256 * HID + TPB - 1) / TPB, TPB, 0, stream>>>(upd_w1,  w1f_upd,  256, 256);
  k_wfrag<<<(128 * HID + TPB - 1) / TPB, TPB, 0, stream>>>(upd_w2,  w2f_upd,  128, 128);

  // 4. five spatial message scales
  const float cuts[5] = {5.0f, 10.0f, 15.0f, 20.0f, 25.0f};
  for (int j = 0; j < 5; ++j) {
    const int* erow = (const int*)d_in[19 + 2 * j];
    const int* ecol = (const int*)d_in[20 + 2 * j];
    int E = in_sizes[19 + 2 * j];
    if (E <= 0) continue;
    int tiles = (E + 15) / 16;
    k_msg<<<tiles, 128, 0, stream>>>(
        xh, pos, erow, ecol, E,
        w1f_msg + (size_t)j * 320 * HID, msg_b1 + (size_t)j * HID,
        w2f_msg + (size_t)j * 128 * HID, msg_b2 + (size_t)j * HID,
        accs + (size_t)j * NNODES * HID, cnts + (size_t)j * NNODES, cuts[j]);
  }
  // 5. bond scale (index 5)
  {
    const int* brow = (const int*)d_in[29];
    const int* bcol = (const int*)d_in[30];
    int E = in_sizes[29];
    if (E > 0) {
      int tiles = (E + 15) / 16;
      k_bond<<<tiles, 128, 0, stream>>>(
          xh, battr, brow, bcol, E,
          w1f_bond, bond_b1, w2f_bond, bond_b2,
          accs + (size_t)5 * NNODES * HID, cnts + (size_t)5 * NNODES);
    }
  }
  // 6. scatter-mean finalize
  {
    int total = 6 * NNODES * HID;
    k_final<<<(total + TPB - 1) / TPB, TPB, 0, stream>>>(accs, cnts, total);
  }
  // 7. fused attention + residual update
  k_attn_upd<<<NNODES / 16, 128, 0, stream>>>(
      x, xh, accs,
      w1f_attn, attn_b1, attn_w2, attn_b2,
      w1f_upd, upd_b1, w2f_upd, upd_b2,
      (float*)d_out);
}
